// IMultiHeadAttention2D_72138270704049
// MI455X (gfx1250) — compile-verified, matrix operand order checked by reading, not run on hardware
//
#include <hip/hip_runtime.h>
#include <hip/hip_bf16.h>

#define N_SP 2304   // 48*48 spatial positions
#define CDIM 256
#define EDIM 512

typedef __attribute__((ext_vector_type(16))) _Float16 v16h;
typedef __attribute__((ext_vector_type(8)))  _Float16 v8h;
typedef __attribute__((ext_vector_type(8)))  float    v8f;

// ---- WMMA helper: D = A(16x32 f16) * B(32x16 f16) + C(16x16 f32) ----
__device__ __forceinline__ v8f wmma32(v16h a, v16h b, v8f c) {
    return __builtin_amdgcn_wmma_f32_16x16x32_f16(
        /*neg_a=*/false, a, /*neg_b=*/false, b,
        /*c_mod=*/(short)0, c, /*reuse_a=*/false, /*reuse_b=*/false);
}

// ---- Fragment load from LDS staged as [row][K] (K contiguous) ----
// A-frag: row = M (lane&15), K chunks {0..7,16..23} (+8 for lanes>=16)
// B-frag: row = N (lane&15), same K chunking -> identical loader.
__device__ __forceinline__ v16h frag_ld(const _Float16* row_base, int stride_h, int kofs) {
    const int lane = threadIdx.x & 31;
    const int row  = lane & 15;
    const int hi   = (lane >> 4) << 3;          // 0 or 8
    const _Float16* p = row_base + row * stride_h + kofs + hi;
    v8h lo8 = *(const v8h*)(p);
    v8h hi8 = *(const v8h*)(p + 16);
    v16h r;
#pragma unroll
    for (int i = 0; i < 8; ++i) { r[i] = lo8[i]; r[i + 8] = hi8[i]; }
    return r;
}

// =====================================================================
// Kernel 1: qkv[b][o][n] = sum_c Wqkv[o][c] * x[b][c][n]   (f32 -> f16)
// Block tile 64(M=o) x 64(N=n), K-step 32, 8 waves (4 Mt x 2 Nt-pairs)
// =====================================================================
__global__ __launch_bounds__(256) void qkv_kernel(
    const float* __restrict__ x, const float* __restrict__ Wqkv,
    _Float16* __restrict__ qkvh) {
    __shared__ alignas(16) _Float16 As[64 * 48];
    __shared__ alignas(16) _Float16 Bs[64 * 48];
    const int n0 = blockIdx.x * 64;
    const int o0 = blockIdx.y * 64;
    const int b  = blockIdx.z;
    const int t  = threadIdx.x;
    const int w  = t >> 5;
    const int mt = w & 3;                 // M-subtile 0..3
    const int np = (w >> 2) * 2;          // N-subtile pair base 0 or 2
    v8f acc0 = {}, acc1 = {};
    const float* xb = x + (size_t)b * CDIM * N_SP;

    for (int k0 = 0; k0 < CDIM; k0 += 32) {
        // A: Wqkv rows [o0,o0+64), cols [k0,k0+32) -> As[row][c] (convert)
#pragma unroll
        for (int r = 0; r < 8; ++r) {
            int idx = t + r * 256;
            int row = idx >> 5, col = idx & 31;
            As[row * 48 + col] = (_Float16)Wqkv[(size_t)(o0 + row) * CDIM + k0 + col];
        }
        // B: x rows [k0,k0+32), cols [n0,n0+64) -> transposed Bs[n][c]
#pragma unroll
        for (int r = 0; r < 8; ++r) {
            int idx = t + r * 256;
            int crow = idx >> 6, ncol = idx & 63;
            Bs[ncol * 48 + crow] = (_Float16)xb[(size_t)(k0 + crow) * N_SP + n0 + ncol];
        }
        __syncthreads();
        v16h a  = frag_ld(As + mt * 16 * 48, 48, 0);
        v16h b0 = frag_ld(Bs + np * 16 * 48, 48, 0);
        v16h b1 = frag_ld(Bs + (np + 1) * 16 * 48, 48, 0);
        acc0 = wmma32(a, b0, acc0);
        acc1 = wmma32(a, b1, acc1);
        __syncthreads();
    }
    // D layout: lane&15 = N, VGPR r -> M = r + 8*(lane>=16)
    const int lane = t & 31;
    const int nl   = lane & 15;
    const int mo   = (lane >> 4) * 8;
    _Float16* ob = qkvh + (size_t)b * 1536 * N_SP;
#pragma unroll
    for (int r = 0; r < 8; ++r) {
        int o = o0 + mt * 16 + mo + r;
        ob[(size_t)o * N_SP + n0 + np * 16 + nl]       = (_Float16)acc0[r];
        ob[(size_t)o * N_SP + n0 + (np + 1) * 16 + nl] = (_Float16)acc1[r];
    }
}

// =====================================================================
// Kernel 2: flash attention per (b,h).  q/k/v stored [d][n] f16.
// Block: 128 query rows (8 waves x 16), key blocks of 128 (BK=128),
// online softmax, 32 WMMAs per wave per key block.
// Y[b][e=h*64+d][n] = softmax(q^T k / 8) v   (f16 out)
// =====================================================================
__global__ __launch_bounds__(256) void attn_kernel(
    const _Float16* __restrict__ qkvh, _Float16* __restrict__ Yh) {
    __shared__ alignas(16) _Float16 Qs[128 * 72];        // [i][d]
    __shared__ alignas(16) _Float16 Ks[128 * 72];        // [j][d]
    __shared__ alignas(16) _Float16 Vs[64 * 136];        // [d][j]
    __shared__ alignas(16) _Float16 Ps[8 * 16 * 136];    // per-wave [i][j]
    const int n0 = blockIdx.x * 128;
    const int h  = blockIdx.y;
    const int b  = blockIdx.z;
    const int t  = threadIdx.x;
    const int w  = t >> 5;
    const int lane = t & 31;

    const _Float16* qb = qkvh + ((size_t)b * 1536 + h * 64) * N_SP;
    const _Float16* kb = qb + (size_t)512  * N_SP;
    const _Float16* vb = qb + (size_t)1024 * N_SP;

    // Stage Q transposed: Qs[i][d] = q[d][n0+i]
    for (int r = 0; r < 32; ++r) {
        int idx = t + r * 256;                 // 128*64 = 8192
        int d = idx >> 7, il = idx & 127;
        Qs[il * 72 + d] = qb[(size_t)d * N_SP + n0 + il];
    }
    __syncthreads();

    // Loop-invariant Q fragments (K-dim = d = 64)
    const _Float16* Qw = Qs + w * 16 * 72;
    v16h qa0 = frag_ld(Qw, 72, 0);
    v16h qa1 = frag_ld(Qw, 72, 32);

    v8f acc[4] = {};                            // O accum, 4 d-subtiles
    float mrow[8], lrow[8];
#pragma unroll
    for (int r = 0; r < 8; ++r) { mrow[r] = -1e30f; lrow[r] = 0.f; }

    _Float16* Pw = Ps + w * 16 * 136;           // per-wave P scratch

    for (int j0 = 0; j0 < N_SP; j0 += 128) {
        // Stage K transposed (Ks[j][d]) and V direct (Vs[d][j])
#pragma unroll
        for (int r = 0; r < 32; ++r) {
            int idx = t + r * 256;             // 64*128 = 8192
            int d = idx >> 7, jl = idx & 127;
            _Float16 kv = kb[(size_t)d * N_SP + j0 + jl];
            _Float16 vv = vb[(size_t)d * N_SP + j0 + jl];
            Ks[jl * 72 + d] = kv;
            Vs[d * 136 + jl] = vv;
        }
        __syncthreads();

        // S = Q K^T  (8 j-subtiles, 2 WMMA each)
        v8f s[8];
#pragma unroll
        for (int jt = 0; jt < 8; ++jt) {
            v16h kb0 = frag_ld(Ks + jt * 16 * 72, 72, 0);
            v16h kb1 = frag_ld(Ks + jt * 16 * 72, 72, 32);
            v8f z = {};
            s[jt] = wmma32(qa1, kb1, wmma32(qa0, kb0, z));
        }
#pragma unroll
        for (int jt = 0; jt < 8; ++jt)
#pragma unroll
            for (int r = 0; r < 8; ++r) s[jt][r] *= 0.125f;

        // Online softmax: rows live across 16 lanes of a half-wave
#pragma unroll
        for (int r = 0; r < 8; ++r) {
            float m = s[0][r];
#pragma unroll
            for (int jt = 1; jt < 8; ++jt) m = fmaxf(m, s[jt][r]);
            m = fmaxf(m, __shfl_xor(m, 1, 32));
            m = fmaxf(m, __shfl_xor(m, 2, 32));
            m = fmaxf(m, __shfl_xor(m, 4, 32));
            m = fmaxf(m, __shfl_xor(m, 8, 32));
            float mn = fmaxf(mrow[r], m);
            float corr = __expf(mrow[r] - mn);
            mrow[r] = mn;
            float psum = 0.f;
#pragma unroll
            for (int jt = 0; jt < 8; ++jt) {
                float p = __expf(s[jt][r] - mn);
                s[jt][r] = p;
                psum += p;
            }
            psum += __shfl_xor(psum, 1, 32);
            psum += __shfl_xor(psum, 2, 32);
            psum += __shfl_xor(psum, 4, 32);
            psum += __shfl_xor(psum, 8, 32);
            lrow[r] = lrow[r] * corr + psum;
#pragma unroll
            for (int dt = 0; dt < 4; ++dt) acc[dt][r] *= corr;
        }

        // P (C/D layout, f32) -> per-wave LDS [i][j] f16 (A-frag source)
        {
            const int nc = lane & 15;
            const int mo = (lane >> 4) * 8;
#pragma unroll
            for (int jt = 0; jt < 8; ++jt)
#pragma unroll
                for (int r = 0; r < 8; ++r)
                    Pw[(mo + r) * 136 + jt * 16 + nc] = (_Float16)s[jt][r];
        }
        // O += P V   (K-dim = j = 128 -> 4 WMMA per d-subtile)
        v16h pa0 = frag_ld(Pw, 136, 0);
        v16h pa1 = frag_ld(Pw, 136, 32);
        v16h pa2 = frag_ld(Pw, 136, 64);
        v16h pa3 = frag_ld(Pw, 136, 96);
#pragma unroll
        for (int dt = 0; dt < 4; ++dt) {
            const _Float16* Vt = Vs + dt * 16 * 136;
            v8f o = acc[dt];
            o = wmma32(pa0, frag_ld(Vt, 136, 0),  o);
            o = wmma32(pa1, frag_ld(Vt, 136, 32), o);
            o = wmma32(pa2, frag_ld(Vt, 136, 64), o);
            o = wmma32(pa3, frag_ld(Vt, 136, 96), o);
            acc[dt] = o;
        }
        __syncthreads();
    }

    // Normalize and store Y[b][e][n] f16 (n contiguous -> one b128 per tile)
    _Float16* Yb = Yh + (size_t)b * EDIM * N_SP;
    const int dl = lane & 15;
    const int io = (lane >> 4) * 8;
#pragma unroll
    for (int dt = 0; dt < 4; ++dt) {
        int e = h * 64 + dt * 16 + dl;
        v8h pk;
#pragma unroll
        for (int r = 0; r < 8; ++r) pk[r] = (_Float16)(acc[dt][r] / lrow[r]);
        *(v8h*)(Yb + (size_t)e * N_SP + n0 + w * 16 + io) = pk;
    }
}

// =====================================================================
// Kernel 3: out[b][c][n] = sum_e Wout[c][e] * Y[b][e][n] + bout[c] (f32)
// =====================================================================
__global__ __launch_bounds__(256) void proj_kernel(
    const _Float16* __restrict__ Yh, const float* __restrict__ Wout,
    const float* __restrict__ bout, float* __restrict__ out) {
    __shared__ alignas(16) _Float16 As[64 * 48];
    __shared__ alignas(16) _Float16 Bs[64 * 48];
    const int n0 = blockIdx.x * 64;
    const int c0 = blockIdx.y * 64;
    const int b  = blockIdx.z;
    const int t  = threadIdx.x;
    const int w  = t >> 5;
    const int mt = w & 3;
    const int np = (w >> 2) * 2;
    v8f acc0 = {}, acc1 = {};
    const _Float16* Yb = Yh + (size_t)b * EDIM * N_SP;

    for (int k0 = 0; k0 < EDIM; k0 += 32) {
#pragma unroll
        for (int r = 0; r < 8; ++r) {
            int idx = t + r * 256;
            int row = idx >> 5, col = idx & 31;
            As[row * 48 + col] = (_Float16)Wout[(size_t)(c0 + row) * EDIM + k0 + col];
        }
#pragma unroll
        for (int r = 0; r < 8; ++r) {
            int idx = t + r * 256;
            int er = idx >> 6, nl = idx & 63;
            Bs[nl * 48 + er] = Yb[(size_t)(k0 + er) * N_SP + n0 + nl];
        }
        __syncthreads();
        v16h a  = frag_ld(As + mt * 16 * 48, 48, 0);
        v16h b0 = frag_ld(Bs + np * 16 * 48, 48, 0);
        v16h b1 = frag_ld(Bs + (np + 1) * 16 * 48, 48, 0);
        acc0 = wmma32(a, b0, acc0);
        acc1 = wmma32(a, b1, acc1);
        __syncthreads();
    }
    const int lane = t & 31;
    const int nl = lane & 15;
    const int mo = (lane >> 4) * 8;
    float* ob = out + (size_t)b * CDIM * N_SP;
#pragma unroll
    for (int r = 0; r < 8; ++r) {
        int c = c0 + mt * 16 + mo + r;
        float bv = bout[c];
        ob[(size_t)c * N_SP + n0 + np * 16 + nl]       = acc0[r] + bv;
        ob[(size_t)c * N_SP + n0 + (np + 1) * 16 + nl] = acc1[r] + bv;
    }
}

extern "C" void kernel_launch(void* const* d_in, const int* in_sizes, int n_in,
                              void* d_out, int out_size, void* d_ws, size_t ws_size,
                              hipStream_t stream) {
    const float* x    = (const float*)d_in[0];   // (4,256,48,48)
    const float* Wqkv = (const float*)d_in[1];   // (1536,256)
    const float* Wout = (const float*)d_in[2];   // (256,512)
    const float* bout = (const float*)d_in[3];   // (256,)
    float* out = (float*)d_out;                  // (4,256,48,48)

    _Float16* qkvh = (_Float16*)d_ws;                         // 4*1536*2304 f16
    _Float16* Yh   = qkvh + (size_t)4 * 1536 * N_SP;          // 4*512*2304 f16

    dim3 blk(256);
    qkv_kernel <<<dim3(N_SP / 64, 1536 / 64, 4), blk, 0, stream>>>(x, Wqkv, qkvh);
    attn_kernel<<<dim3(N_SP / 128, 8, 4),        blk, 0, stream>>>(qkvh, Yh);
    proj_kernel<<<dim3(N_SP / 64, CDIM / 64, 4), blk, 0, stream>>>(Yh, Wout, bout, out);
}
